// GRF_HGNN_K4_15659450761596
// MI455X (gfx1250) — matrix operand order repaired and checked by reading, not compiled, not run on hardware
//
#include <hip/hip_runtime.h>
#include <hip/hip_bf16.h>
#include <stdint.h>

// ---------------- problem constants (match reference) ----------------
#define H      128
#define LAYERS 4
#define NB     131072
#define NJ     393216
#define NF     131072
#define E_BJ   131072
#define E_JB   131072
#define E_JJ   262144
#define E_JF   131072
#define E_FJ   131072
#define E_GT   196608
#define E_GS   196608

// Harness contract: "integer -> const int*"
typedef int eidx_t;

typedef __attribute__((ext_vector_type(16))) __bf16 v16bf;
typedef __attribute__((ext_vector_type(8)))  float  v8f;

union BPack {
  v16bf v;
  uint4 q[2];
};

__device__ __forceinline__ unsigned short f2bf(float f) {
  unsigned int u = __float_as_uint(f);
  u += 0x7FFFu + ((u >> 16) & 1u);         // round-to-nearest-even
  return (unsigned short)(u >> 16);
}

// ---------------------------------------------------------------------
// Pack a 128x128 fp32 weight (optionally a sum of up to 3) into the
// per-lane bf16 WMMA B-matrix layout:
//   dst[((kc*8 + nt)*32 + lane)*16 + j]  holds  B[k,n] = W[n,k]
//   with k = kc*32 + (lane>>4)*16 + j,  n = nt*16 + (lane&15)
// so a GEMM lane reads 32 contiguous bytes per (kc,nt) tile.
// ---------------------------------------------------------------------
__global__ void pack_b128(const float* __restrict__ W0,
                          const float* __restrict__ W1,
                          const float* __restrict__ W2,
                          unsigned short* __restrict__ dst) {
  int idx = blockIdx.x * blockDim.x + threadIdx.x;   // 0..16383
  if (idx >= H * H) return;
  int j    = idx & 15;
  int lane = (idx >> 4) & 31;
  int nt   = (idx >> 9) & 7;
  int kc   = idx >> 12;
  int k = kc * 32 + (lane >> 4) * 16 + j;
  int n = nt * 16 + (lane & 15);
  float v = W0[n * H + k];
  if (W1) v += W1[n * H + k];
  if (W2) v += W2[n * H + k];
  dst[idx] = f2bf(v);
}

// Per-layer summed biases for the fused HeteroConv terms (root + rel biases).
__global__ void bias_sums_k(const float* __restrict__ brel,   // [L,7,H]
                            float* __restrict__ outJ,         // [L,H] et 0+2+4
                            float* __restrict__ outB) {       // [L,H] et 1+5+6
  int i = blockIdx.x * blockDim.x + threadIdx.x;
  if (i >= LAYERS * H) return;
  int l = i >> 7, hh = i & 127;
  const float* base = brel + (size_t)l * 7 * H;
  outJ[i] = base[0 * H + hh] + base[2 * H + hh] + base[4 * H + hh];
  outB[i] = base[1 * H + hh] + base[5 * H + hh] + base[6 * H + hh];
}

// ---------------------------------------------------------------------
// GEMM: Cout[m,n] = (Cin?Cin[m,n]:0) + sum_k A[m,k]*W[n,k] + bias[n] (opt ReLU)
// N = K = 128 fixed.  The packed bf16 weight (32 KB) is staged once per
// block into LDS and shared by 8 waves x several row tiles (grid-stride).
// One wave computes a 16(M) x 128(N) strip: 4 K-chunks x 8 N-tiles = 32 WMMA.
// All M are multiples of 128 -> no tails, EXEC all-ones (WMMA requirement).
// ---------------------------------------------------------------------
__global__ void __launch_bounds__(256)
gemm128(const float* __restrict__ A, const unsigned short* __restrict__ Bpk,
        const float* __restrict__ bias, const float* __restrict__ Cin,
        float* __restrict__ Cout, int Mtiles, int relu) {
  __shared__ unsigned short Bsh[H * H];            // 32 KB
  {
    const uint4* src = (const uint4*)Bpk;
    uint4* dst = (uint4*)Bsh;
    for (int i = threadIdx.x; i < (H * H) / 8; i += 256) dst[i] = src[i];
  }
  __syncthreads();

  const int wave  = threadIdx.x >> 5;
  const int lane  = threadIdx.x & 31;
  const int khalf = lane >> 4;                     // 0|1: K half per lane group
  const int mrow0 = khalf * 8;                     // C/D row base for this lane
  const int ncol  = lane & 15;

  for (int rt = blockIdx.x * 8 + wave; rt < Mtiles; rt += gridDim.x * 8) {
    const int m = rt * 16 + (lane & 15);           // A row for this lane

    v8f acc[8];
#pragma unroll
    for (int nt = 0; nt < 8; ++nt) {
      if (Cin) {
#pragma unroll
        for (int r = 0; r < 8; ++r)
          acc[nt][r] = Cin[(size_t)(rt * 16 + mrow0 + r) * H + nt * 16 + ncol];
      } else {
#pragma unroll
        for (int r = 0; r < 8; ++r) acc[nt][r] = 0.0f;
      }
    }

#pragma unroll
    for (int kc = 0; kc < 4; ++kc) {
      // A 16x32 bf16 tile. lane<16: K = 0..15; lane>=16: K = 16..31.
      const float* arow = A + (size_t)m * H + kc * 32 + khalf * 16;
      const float4 f0 = *(const float4*)(arow);
      const float4 f1 = *(const float4*)(arow + 4);
      const float4 f2 = *(const float4*)(arow + 8);
      const float4 f3 = *(const float4*)(arow + 12);
      v16bf av;
      av[0]  = (__bf16)f0.x; av[1]  = (__bf16)f0.y; av[2]  = (__bf16)f0.z; av[3]  = (__bf16)f0.w;
      av[4]  = (__bf16)f1.x; av[5]  = (__bf16)f1.y; av[6]  = (__bf16)f1.z; av[7]  = (__bf16)f1.w;
      av[8]  = (__bf16)f2.x; av[9]  = (__bf16)f2.y; av[10] = (__bf16)f2.z; av[11] = (__bf16)f2.w;
      av[12] = (__bf16)f3.x; av[13] = (__bf16)f3.y; av[14] = (__bf16)f3.z; av[15] = (__bf16)f3.w;
#pragma unroll
      for (int nt = 0; nt < 8; ++nt) {
        BPack b;
        const uint4* bp =
            (const uint4*)(Bsh + ((kc * 8 + nt) * 32 + lane) * 16);
        b.q[0] = bp[0];
        b.q[1] = bp[1];
        acc[nt] = __builtin_amdgcn_wmma_f32_16x16x32_bf16(
            false, av, false, b.v, (short)0, acc[nt], false, false);
      }
    }

#pragma unroll
    for (int nt = 0; nt < 8; ++nt) {
      const float bv = bias ? bias[nt * 16 + ncol] : 0.0f;
#pragma unroll
      for (int r = 0; r < 8; ++r) {
        float v = acc[nt][r] + bv;
        if (relu) v = fmaxf(v, 0.0f);
        Cout[(size_t)(rt * 16 + mrow0 + r) * H + nt * 16 + ncol] = v;
      }
    }
  }
}

// ----------------- sparse message passing (HBM-bound) -----------------
// Transform-before-scatter: tmp = h_src @ Wrel.T is done densely first;
// here we just gather tmp[src] and atomically accumulate into dst rows.
// One wave per edge: each lane moves 4 channels (float4 gather + 4 atomics).
__global__ void scatter_add_k(const float* __restrict__ msg,
                              const eidx_t* __restrict__ ei, int E,
                              float* __restrict__ dst) {
  long long gid = (long long)blockIdx.x * blockDim.x + threadIdx.x;
  int e = (int)(gid >> 5);
  int lane = (int)(gid & 31);
  if (e >= E) return;
  int s = ei[e];
  int d = ei[E + e];
  const float4 v = *(const float4*)(msg + (size_t)s * H + lane * 4);
  float* out = dst + (size_t)d * H + lane * 4;
  atomicAdd(out + 0, v.x);
  atomicAdd(out + 1, v.y);
  atomicAdd(out + 2, v.z);
  atomicAdd(out + 3, v.w);
}

__global__ void count_k(const eidx_t* __restrict__ ei, int E,
                        float* __restrict__ cnt) {
  int e = blockIdx.x * blockDim.x + threadIdx.x;
  if (e < E) atomicAdd(&cnt[ei[E + e]], 1.0f);
}

// dst[i] += src[i] / max(cnt[row],1)   (mean-aggregated edge types)
__global__ void scale_add_k(float* __restrict__ dst, const float* __restrict__ src,
                            const float* __restrict__ cnt, int total) {
  int i = blockIdx.x * blockDim.x + threadIdx.x;
  if (i < total) dst[i] += src[i] / fmaxf(cnt[i >> 7], 1.0f);
}

__global__ void relu_res_k(float* __restrict__ h, const float* __restrict__ nw,
                           int total) {
  int i = blockIdx.x * blockDim.x + threadIdx.x;
  if (i < total) h[i] += fmaxf(nw[i], 0.0f);
}

// ------------------------- encoder / decoder --------------------------
__global__ void encode_k(const float* __restrict__ x, const float* __restrict__ W,
                         const float* __restrict__ b, float* __restrict__ h, int F) {
  __shared__ float xs[16];
  const int node = blockIdx.x;
  const int t = threadIdx.x;         // output channel, blockDim = 128
  if (t < F) xs[t] = x[(size_t)node * F + t];
  __syncthreads();
  float acc = b[t];
  for (int f = 0; f < F; ++f) acc += W[t * F + f] * xs[f];
  h[(size_t)node * H + t] = fmaxf(acc, 0.0f);
}

__global__ void decode_k(const float* __restrict__ hf, const float* __restrict__ W,
                         const float* __restrict__ b, float* __restrict__ out) {
  int gid = blockIdx.x * blockDim.x + threadIdx.x;
  int n = gid >> 5, lane = gid & 31;
  const float4 v = *(const float4*)(hf + (size_t)n * H + lane * 4);
  const float4 w = *(const float4*)(W + lane * 4);
  float s = v.x * w.x + v.y * w.y + v.z * w.z + v.w * w.w;
  for (int o = 16; o > 0; o >>= 1) s += __shfl_xor(s, o, 32);
  if (lane == 0) out[n] = s + b[0];
}

// ---------------------------------------------------------------------
extern "C" void kernel_launch(void* const* d_in, const int* in_sizes, int n_in,
                              void* d_out, int out_size, void* d_ws, size_t ws_size,
                              hipStream_t stream) {
  const float* x_base   = (const float*)d_in[0];
  const float* x_joint  = (const float*)d_in[1];
  const float* x_foot   = (const float*)d_in[2];
  const float* eWb      = (const float*)d_in[3];
  const float* ebb      = (const float*)d_in[4];
  const float* eWj      = (const float*)d_in[5];
  const float* ebj      = (const float*)d_in[6];
  const float* eWf      = (const float*)d_in[7];
  const float* ebf      = (const float*)d_in[8];
  const float* Wrel     = (const float*)d_in[9];    // [L,7,H,H]
  const float* brel     = (const float*)d_in[10];   // [L,7,H]
  const float* Wroot    = (const float*)d_in[11];   // [L,7,H,H]
  const float* btW1     = (const float*)d_in[12];
  const float* btb1     = (const float*)d_in[13];
  const float* btW2     = (const float*)d_in[14];
  const float* btb2     = (const float*)d_in[15];
  const float* decW     = (const float*)d_in[16];
  const float* decb     = (const float*)d_in[17];
  const eidx_t* ei_bj   = (const eidx_t*)d_in[18];
  const eidx_t* ei_jb   = (const eidx_t*)d_in[19];
  const eidx_t* ei_jj   = (const eidx_t*)d_in[20];
  const eidx_t* ei_jf   = (const eidx_t*)d_in[21];
  const eidx_t* ei_fj   = (const eidx_t*)d_in[22];
  const eidx_t* ei_gt   = (const eidx_t*)d_in[23];
  const eidx_t* ei_gs   = (const eidx_t*)d_in[24];
  float* out = (float*)d_out;

  // ---- workspace carve-up (bump allocator, 256B aligned) ----
  char* ws = (char*)d_ws;
  size_t off = 0;
  auto carve = [&](size_t bytes) -> char* {
    char* p = ws + off;
    off = (off + bytes + 255) & ~(size_t)255;
    return p;
  };
  const size_t MATW = (size_t)H * H;                 // 16384 elems
  float* hb   = (float*)carve((size_t)NB * H * 4);
  float* hj   = (float*)carve((size_t)NJ * H * 4);
  float* hf   = (float*)carve((size_t)NF * H * 4);
  float* nwb  = (float*)carve((size_t)NB * H * 4);
  float* nwj  = (float*)carve((size_t)NJ * H * 4);
  float* nwf  = (float*)carve((size_t)NF * H * 4);
  float* tsrc = (float*)carve((size_t)NJ * H * 4);   // transformed messages / MLP tmp
  float* aggB = (float*)carve((size_t)NB * H * 4);   // mean-aggr side buffer
  float* cntT = (float*)carve((size_t)NB * 4);       // gt degree counts
  float* cntS = (float*)carve((size_t)NB * 4);       // gs degree counts
  unsigned short* wrel_pk   = (unsigned short*)carve(LAYERS * 7 * MATW * 2);
  unsigned short* wrootJ_pk = (unsigned short*)carve(LAYERS * MATW * 2);
  unsigned short* wrootB_pk = (unsigned short*)carve(LAYERS * MATW * 2);
  unsigned short* wrootF_pk = (unsigned short*)carve(LAYERS * MATW * 2);
  unsigned short* w1_pk     = (unsigned short*)carve(MATW * 2);
  unsigned short* w2_pk     = (unsigned short*)carve(MATW * 2);
  float* bsJ = (float*)carve(LAYERS * H * 4);
  float* bsB = (float*)carve(LAYERS * H * 4);
  (void)ws_size; (void)in_sizes; (void)n_in; (void)out_size;

  auto zero = [&](void* p, size_t b) { hipMemsetAsync(p, 0, b, stream); };
  auto gemm = [&](const float* A, const unsigned short* B, const float* bias,
                  const float* Cin, float* Cout, int M, int relu) {
    int mt = M / 16;
    int grid = mt / 8;
    if (grid > 1024) grid = 1024;
    gemm128<<<grid, 256, 0, stream>>>(A, B, bias, Cin, Cout, mt, relu);
  };
  auto scat = [&](const float* msg, const eidx_t* ei, int E, float* dst) {
    scatter_add_k<<<E / 8, 256, 0, stream>>>(msg, ei, E, dst);
  };

  // ---- weight prep (deterministic, every launch) ----
  for (int l = 0; l < LAYERS; ++l) {
    for (int et = 0; et < 7; ++et)
      pack_b128<<<64, 256, 0, stream>>>(Wrel + ((size_t)l * 7 + et) * MATW,
                                        nullptr, nullptr,
                                        wrel_pk + ((size_t)l * 7 + et) * MATW);
    const float* Wo = Wroot + (size_t)l * 7 * MATW;
    pack_b128<<<64, 256, 0, stream>>>(Wo + 0 * MATW, Wo + 2 * MATW, Wo + 4 * MATW,
                                      wrootJ_pk + (size_t)l * MATW);
    pack_b128<<<64, 256, 0, stream>>>(Wo + 1 * MATW, Wo + 5 * MATW, Wo + 6 * MATW,
                                      wrootB_pk + (size_t)l * MATW);
    pack_b128<<<64, 256, 0, stream>>>(Wo + 3 * MATW, nullptr, nullptr,
                                      wrootF_pk + (size_t)l * MATW);
  }
  pack_b128<<<64, 256, 0, stream>>>(btW1, nullptr, nullptr, w1_pk);
  pack_b128<<<64, 256, 0, stream>>>(btW2, nullptr, nullptr, w2_pk);
  bias_sums_k<<<2, 256, 0, stream>>>(brel, bsJ, bsB);

  // Degree counts for mean-aggregated edge types depend only on the graph:
  // compute once, reuse every layer.
  zero(cntT, (size_t)NB * 4);
  zero(cntS, (size_t)NB * 4);
  count_k<<<E_GT / 256, 256, 0, stream>>>(ei_gt, E_GT, cntT);
  count_k<<<E_GS / 256, 256, 0, stream>>>(ei_gs, E_GS, cntS);

  // ---- encoders ----
  encode_k<<<NB, H, 0, stream>>>(x_base, eWb, ebb, hb, 12);
  encode_k<<<NJ, H, 0, stream>>>(x_joint, eWj, ebj, hj, 3);
  encode_k<<<NF, H, 0, stream>>>(x_foot, eWf, ebf, hf, 2);

  // ---- layers: transform-before-scatter GraphConv ----
  for (int l = 0; l < LAYERS; ++l) {
    const unsigned short* Wr = wrel_pk + (size_t)l * 7 * MATW;

    // joint dst: nwj = hj@sum(Wroot).T + sum(brel); then += scattered W*h_src
    gemm(hj, wrootJ_pk + (size_t)l * MATW, bsJ + l * H, nullptr, nwj, NJ, 0);
    gemm(hb, Wr + 0 * MATW, nullptr, nullptr, tsrc, NB, 0);   // bj
    scat(tsrc, ei_bj, E_BJ, nwj);
    gemm(hj, Wr + 2 * MATW, nullptr, nullptr, tsrc, NJ, 0);   // jj
    scat(tsrc, ei_jj, E_JJ, nwj);
    gemm(hf, Wr + 4 * MATW, nullptr, nullptr, tsrc, NF, 0);   // fj
    scat(tsrc, ei_fj, E_FJ, nwj);

    // base dst
    gemm(hb, wrootB_pk + (size_t)l * MATW, bsB + l * H, nullptr, nwb, NB, 0);
    gemm(hj, Wr + 1 * MATW, nullptr, nullptr, tsrc, NJ, 0);   // jb
    scat(tsrc, ei_jb, E_JB, nwb);
    // gt (mean): mean/linear commute; 1/deg applied before merging into nwb
    gemm(hb, Wr + 5 * MATW, nullptr, nullptr, tsrc, NB, 0);
    zero(aggB, (size_t)NB * H * 4);
    scat(tsrc, ei_gt, E_GT, aggB);
    scale_add_k<<<NB * H / 256, 256, 0, stream>>>(nwb, aggB, cntT, NB * H);
    // gs (mean)
    gemm(hb, Wr + 6 * MATW, nullptr, nullptr, tsrc, NB, 0);
    zero(aggB, (size_t)NB * H * 4);
    scat(tsrc, ei_gs, E_GS, aggB);
    scale_add_k<<<NB * H / 256, 256, 0, stream>>>(nwb, aggB, cntS, NB * H);

    // foot dst
    gemm(hf, wrootF_pk + (size_t)l * MATW, brel + ((size_t)l * 7 + 3) * H,
         nullptr, nwf, NF, 0);
    gemm(hj, Wr + 3 * MATW, nullptr, nullptr, tsrc, NJ, 0);   // jf
    scat(tsrc, ei_jf, E_JF, nwf);

    // base transform MLP + residual fused into second GEMM (tsrc as tmp)
    gemm(nwb, w1_pk, btb1, nullptr, tsrc, NB, 1);             // tmp = relu(nwb@W1+b1)
    gemm(tsrc, w2_pk, btb2, hb, hb, NB, 0);                   // hb += tmp@W2 + b2

    // joint/foot residuals
    relu_res_k<<<NJ * H / 256, 256, 0, stream>>>(hj, nwj, NJ * H);
    relu_res_k<<<NF * H / 256, 256, 0, stream>>>(hf, nwf, NF * H);
  }

  // ---- decoder ----
  decode_k<<<NF * 32 / 256, 256, 0, stream>>>(hf, decW, decb, out);
}
